// NeuralPDA_5720896438735
// MI455X (gfx1250) — matrix operand order repaired and checked by reading, not compiled, not run on hardware
//
#include <hip/hip_runtime.h>

// ---------------------------------------------------------------------------
// NeuralPDA on MI455X (gfx1250): serial RNN-PDA scan, 4 fused kernels per step
// (all GEMMs on v_wmma_f32_16x16x32_f16), logits hoisted into one async-LDS-
// staged, double-buffered, register-blocked 4096x32000 f16 WMMA GEMM.
// ---------------------------------------------------------------------------

#define BB    64        // batch
#define TT    64        // time steps
#define VOCAB 32000
#define DD    512       // d_model
#define NDC   256       // code dim
#define NNT   512       // codebook size
#define CAP   130       // 2*T+2
#define ZDIM  1280      // D + ND + D
#define NCELL 1536      // 3*D (tok | nt | state)
#define ZCODE 1024      // 2*D
#define DECAYF 0.99f

typedef __attribute__((ext_vector_type(16))) _Float16 v16h;
typedef __attribute__((ext_vector_type(8)))  _Float16 v8h;
typedef __attribute__((ext_vector_type(8)))  float    v8f;

// ---- A fragment loader: 16-bit A 16x32, row-major source, lda in halves ----
__device__ __forceinline__ v16h load_afrag(const _Float16* A, int lda, int row0,
                                           int k0, int lane) {
  const int m  = lane & 15;
  const int kb = (lane >> 4) << 4;                 // 0 or 16
  const _Float16* p = A + (size_t)(row0 + m) * lda + k0 + kb;
  v8h lo = *(const v8h*)p;                         // K = k0+kb .. +7
  v8h hi = *(const v8h*)(p + 8);                   // K = k0+kb+8 .. +15
  v16h r;
#pragma unroll
  for (int i = 0; i < 8; ++i) { r[i] = lo[i]; r[8 + i] = hi[i]; }
  return r;
}

// ---- B fragment: packed [kt][nt][lane][16] -> single coalesced v16h load ----
__device__ __forceinline__ v16h load_bfrag(const _Float16* pk, int nt_total,
                                           int kt, int nt, int lane) {
  return *(const v16h*)(pk + (((size_t)kt * nt_total + nt) * 32 + lane) * 16);
}

#define WMMA_F16(a, b, c) \
  __builtin_amdgcn_wmma_f32_16x16x32_f16(false, (a), false, (b), (short)0, (c), false, false)

// ---- CDNA5 async memory->LDS copy (ASYNCcnt path), 16B per lane ----
__device__ __forceinline__ void async_g2l_b128(const _Float16* gsrc, _Float16* ldst) {
  unsigned ldsoff = (unsigned)(uintptr_t)ldst;     // LDS aperture: low 32 bits
  asm volatile("global_load_async_to_lds_b128 %0, %1, off"
               :: "v"(ldsoff), "v"(gsrc) : "memory");
}
__device__ __forceinline__ void wait_asynccnt0() {
  asm volatile("s_wait_asynccnt 0x0" ::: "memory");
}

// ===========================================================================
// Weight packing: f32 row-major [K][N] (or transposed src) -> f16 frag-major
// ===========================================================================
__global__ void k_pack_generic(const float* __restrict__ src, _Float16* __restrict__ dst,
                               int K, int N, int NT, int transposed, int total) {
  int o = blockIdx.x * blockDim.x + threadIdx.x;
  if (o >= total) return;
  int j    = o & 15;
  int lane = (o >> 4) & 31;
  int rest = o >> 9;
  int nt   = rest % NT;
  int kt   = rest / NT;
  int k = kt * 32 + ((lane >> 4) << 4) + j;
  int n = nt * 16 + (lane & 15);
  float v = transposed ? src[(size_t)n * K + k] : src[(size_t)k * N + n];
  dst[o] = (_Float16)v;
}

__global__ void k_pack_cell(const float* __restrict__ Wt, const float* __restrict__ Wn,
                            const float* __restrict__ Ws, _Float16* __restrict__ dst,
                            int total) {
  int o = blockIdx.x * blockDim.x + threadIdx.x;
  if (o >= total) return;
  int j    = o & 15;
  int lane = (o >> 4) & 31;
  int rest = o >> 9;
  int nt   = rest % (NCELL / 16);
  int kt   = rest / (NCELL / 16);
  int k = kt * 32 + ((lane >> 4) << 4) + j;
  int n = nt * 16 + (lane & 15);
  float v;
  if (n < DD)            v = Wt[(size_t)k * DD + n];
  else if (n < 2 * DD)   v = Wn[(size_t)k * DD + (n - DD)];
  else                   v = Ws[(size_t)k * DD + (n - 2 * DD)];
  dst[o] = (_Float16)v;
}

__global__ void k_cbsq(const float* __restrict__ cb, float* __restrict__ cbsq) {
  int n = blockIdx.x * blockDim.x + threadIdx.x;
  if (n >= NNT) return;
  float s = 0.f;
  for (int d = 0; d < NDC; ++d) { float v = cb[(size_t)n * NDC + d]; s += v * v; }
  cbsq[n] = s;
}

__global__ void k_setptr(int* ptr) {
  int b = threadIdx.x;
  if (b < BB) ptr[b] = 1;
}

// ===========================================================================
// Per-step kernel 1: fused z-build + stack pop (one block per batch element)
// ===========================================================================
__global__ void k_buildz_pop(const int* __restrict__ x, const int* __restrict__ xmask,
                             const float* __restrict__ emb, const float* __restrict__ stk,
                             int* __restrict__ ptr, int* __restrict__ smask,
                             const float* __restrict__ hcur, _Float16* __restrict__ z,
                             int t) {
  const int b   = blockIdx.x;
  const int p   = ptr[b];             // all threads of block b read before write
  const int tok = x[b * TT + t];
  for (int j = threadIdx.x; j < ZDIM; j += 256) {
    float v;
    if (j < DD)             v = emb[(size_t)tok * DD + j];
    else if (j < DD + NDC)  v = (p > 0) ? stk[((size_t)b * CAP + (p - 1)) * NDC + (j - DD)]
                                        : 0.f;
    else                    v = hcur[(size_t)b * DD + (j - DD - NDC)];
    z[(size_t)b * ZDIM + j] = (_Float16)v;
  }
  __syncthreads();
  if (threadIdx.x == 0) {             // pop bookkeeping (block-local, race-free)
    int tm   = (p > 0) ? 1 : 0;
    smask[b] = xmask[b * TT + t] * tm;
    ptr[b]   = p - tm;
  }
}

// ===========================================================================
// Per-step kernel 2: fused cell GEMM [64x1280]@[1280x1536], tanh on tok/nt
// ===========================================================================
__global__ void k_cell_gemm(const _Float16* __restrict__ z, const _Float16* __restrict__ wpk,
                            float* __restrict__ hnext, _Float16* __restrict__ zc,
                            _Float16* __restrict__ ht16, int t) {
  const int lane = threadIdx.x & 31;
  const int rt   = threadIdx.x >> 5;     // 4 waves -> 4 row tiles (M=64)
  const int ct   = blockIdx.x;           // 96 col tiles (N=1536)
  v8f c = {};
  for (int kt = 0; kt < ZDIM / 32; ++kt) {
    v16h a = load_afrag(z, ZDIM, rt * 16, kt * 32, lane);
    v16h b = load_bfrag(wpk, NCELL / 16, kt, ct, lane);
    c = WMMA_F16(a, b, c);
  }
  const int n     = ct * 16 + (lane & 15);
  const int rbase = rt * 16 + ((lane >> 4) << 3);
#pragma unroll
  for (int i = 0; i < 8; ++i) {
    int   b = rbase + i;
    float v = c[i];
    if (n < ZCODE) v = tanhf(v);
    if (n < DD) {                                     // h_t
      ht16[((size_t)t * BB + b) * DD + n] = (_Float16)v;
      zc[(size_t)b * ZCODE + DD + n]      = (_Float16)v;
    } else if (n < ZCODE) {                           // h_ntt
      zc[(size_t)b * ZCODE + (n - DD)] = (_Float16)v;
    } else {                                          // h_state (no tanh)
      hnext[(size_t)b * DD + (n - ZCODE)] = v;
    }
  }
}

// ===========================================================================
// Per-step kernel 3: codes GEMM [64x1024]@[1024x512]
// ===========================================================================
__global__ void k_codes_gemm(const _Float16* __restrict__ zc, const _Float16* __restrict__ wpk,
                             float* __restrict__ c32, _Float16* __restrict__ c16,
                             float* __restrict__ out_raw, int t) {
  const int lane = threadIdx.x & 31;
  const int rt   = threadIdx.x >> 5;     // 4 row tiles
  const int ct   = blockIdx.x;           // 32 col tiles
  v8f c = {};
  for (int kt = 0; kt < ZCODE / 32; ++kt) {
    v16h a = load_afrag(zc, ZCODE, rt * 16, kt * 32, lane);
    v16h b = load_bfrag(wpk, 2 * NDC / 16, kt, ct, lane);
    c = WMMA_F16(a, b, c);
  }
  const int n     = ct * 16 + (lane & 15);
  const int rbase = rt * 16 + ((lane >> 4) << 3);
#pragma unroll
  for (int i = 0; i < 8; ++i) {
    int   b = rbase + i;
    float v = c[i];
    c32[(size_t)b * (2 * NDC) + n] = v;
    c16[(size_t)b * (2 * NDC) + n] = (_Float16)v;
    out_raw[((size_t)b * TT + t) * (2 * NDC) + n] = v;   // raw_codes (B,T,2,ND)
  }
}

// ===========================================================================
// Per-step kernel 4: fused VQ = distance WMMA GEMM + argmin + push + EMA.
// One block (256 thr, 8 waves) owns one 16-row tile = 8 complete batches.
// ===========================================================================
__global__ void k_vq(const _Float16* __restrict__ c16, const _Float16* __restrict__ cbpk,
                     const float* __restrict__ c32, const float* __restrict__ cbsq,
                     const float* __restrict__ cb, const int* __restrict__ smask,
                     int* __restrict__ ptr, float* __restrict__ stk,
                     float* __restrict__ accm, float* __restrict__ accn,
                     float* __restrict__ out_push, int t) {
  __shared__ float sdot[16][NNT];     // 32 KB: dot(code_row, cb_n)
  __shared__ float srowsq[16];
  __shared__ int   sqi[16];
  __shared__ int   swi[8][2];
  const int rt   = blockIdx.x;        // 8 row tiles (rows = b*2+k, 128 total)
  const int wave = threadIdx.x >> 5;  // 8 waves
  const int lane = threadIdx.x & 31;

  // --- hoist all A fragments (reused across 4 column tiles per wave) ---
  v16h afr[NDC / 32];
#pragma unroll
  for (int kt = 0; kt < NDC / 32; ++kt)
    afr[kt] = load_afrag(c16, NDC, rt * 16, kt * 32, lane);

  // --- dots via WMMA: wave handles 4 of 32 column tiles ---
#pragma unroll
  for (int cc = 0; cc < 4; ++cc) {
    const int ct = wave * 4 + cc;
    v8f c = {};
#pragma unroll
    for (int kt = 0; kt < NDC / 32; ++kt) {
      v16h b = load_bfrag(cbpk, NNT / 16, kt, ct, lane);
      c = WMMA_F16(afr[kt], b, c);
    }
    const int n  = ct * 16 + (lane & 15);
    const int rl = (lane >> 4) << 3;
#pragma unroll
    for (int i = 0; i < 8; ++i) sdot[rl + i][n] = c[i];
  }

  // --- |code|^2 per row (waves 0..7 handle 2 rows each, wave32 reduction) ---
#pragma unroll
  for (int rr = wave * 2; rr < wave * 2 + 2; ++rr) {
    const int row = rt * 16 + rr;
    float s = 0.f;
    for (int d = lane; d < NDC; d += 32) {
      float v = c32[(size_t)row * NDC + d];
      s += v * v;
    }
#pragma unroll
    for (int off = 16; off > 0; off >>= 1) s += __shfl_xor(s, off, 32);
    if (lane == 0) srowsq[rr] = s;
  }
  __syncthreads();

  // --- argmin_n rowsq - 2*dot + |cb_n|^2 (first-index ties like jnp.argmin) ---
#pragma unroll
  for (int rr = wave * 2; rr < wave * 2 + 2; ++rr) {
    const float rsq = srowsq[rr];
    float bv = 3.0e38f;
    int   bi = 0;
    for (int n = lane; n < NNT; n += 32) {
      float d2 = rsq - 2.0f * sdot[rr][n] + cbsq[n];
      if (d2 < bv) { bv = d2; bi = n; }
    }
#pragma unroll
    for (int off = 16; off > 0; off >>= 1) {
      float ov = __shfl_xor(bv, off, 32);
      int   oi = __shfl_xor(bi, off, 32);
      if (ov < bv || (ov == bv && oi < bi)) { bv = ov; bi = oi; }
    }
    if (lane == 0) sqi[rr] = bi;
  }
  __syncthreads();

  // --- serialized stack-pointer update: threads 0..7 own one batch each ---
  if (threadIdx.x < 8) {
    const int bl = threadIdx.x;
    const int b  = rt * 8 + bl;
    const int sm = smask[b];
    int p = ptr[b];
#pragma unroll
    for (int kk = 0; kk < 2; ++kk) {
      int qi = sqi[bl * 2 + kk];
      int pm = qi * sm;                       // push_mask = qidx * step_mask
      out_push[((size_t)b * TT + t) * 2 + kk] = (float)pm;
      if (pm != 0) { swi[bl][kk] = (p < CAP - 1) ? p : (CAP - 1); p += 1; }
      else         { swi[bl][kk] = -1; }
    }
    ptr[b] = p;
  }
  __syncthreads();

  // --- stack pushes (codebook gather) + EMA accumulation ---
  for (int bl = 0; bl < 8; ++bl) {
    const int b  = rt * 8 + bl;
    const int sm = smask[b];
#pragma unroll
    for (int kk = 0; kk < 2; ++kk) {
      const int qi = sqi[bl * 2 + kk];
      const int w  = swi[bl][kk];
      if (w >= 0)
        for (int d = threadIdx.x; d < NDC; d += 256)
          stk[((size_t)b * CAP + w) * NDC + d] = cb[(size_t)qi * NDC + d];
      if (sm) {
        if (threadIdx.x == 0) atomicAdd(accn + qi, 1.0f);
        for (int d = threadIdx.x; d < NDC; d += 256)
          atomicAdd(accm + (size_t)qi * NDC + d,
                    c32[(size_t)(b * 2 + kk) * NDC + d]);
      }
    }
  }
}

// ===========================================================================
// Logits GEMM: h_t_all[4096x512] @ W_pred[512x32000].
// Block = 8 waves (128 rows) x 64 cols. B-fragments are staged into LDS with
// CDNA5 async copies (global_load_async_to_lds_b128, ASYNCcnt), double-
// buffered so the fetch of K-slice kt+1 overlaps the 4 WMMAs of slice kt.
// Each wave register-blocks 4 N tiles per A-fragment.
// ===========================================================================
__device__ __forceinline__ void store_ctile(const v8f& c, float* __restrict__ out,
                                            int rbase, int n) {
#pragma unroll
  for (int i = 0; i < 8; ++i) {
    int r  = rbase + i;                 // r = t*B + b
    int tt = r >> 6;
    int bb = r & 63;
    out[((size_t)bb * TT + tt) * VOCAB + n] = c[i];
  }
}

__global__ void k_logits(const _Float16* __restrict__ ht16, const _Float16* __restrict__ wpk,
                         float* __restrict__ out_logits) {
  __shared__ __align__(32) _Float16 sB[2][4 * 32 * 16];   // 2 x 4 B-frags = 8 KB
  const int lane = threadIdx.x & 31;
  const int wave = threadIdx.x >> 5;            // 8 waves
  const int rt   = blockIdx.y * 8 + wave;       // 256 row tiles (M=4096)
  const int ctg  = blockIdx.x;                  // 500 groups of 4 col tiles

  // issue one async 16B memory->LDS copy per lane (4 KB per block per slice)
  auto issue = [&](int kt) {
    const _Float16* src =
        wpk + ((size_t)kt * (VOCAB / 16) + (size_t)ctg * 4) * 32 * 16;
    async_g2l_b128(src + threadIdx.x * 8, &sB[kt & 1][threadIdx.x * 8]);
  };

  v8f a0 = {}, a1 = {}, a2 = {}, a3 = {};
  issue(0);
  for (int kt = 0; kt < DD / 32; ++kt) {
    wait_asynccnt0();                 // drain this wave's async copies
    __syncthreads();                  // all waves' copies of slice kt landed
    if (kt + 1 < DD / 32) issue(kt + 1);   // overlap next fetch with compute
    const _Float16* buf = sB[kt & 1];
    v16h a  = load_afrag(ht16, DD, rt * 16, kt * 32, lane);
    v16h b0 = *(const v16h*)(buf + (0 * 32 + lane) * 16);
    v16h b1 = *(const v16h*)(buf + (1 * 32 + lane) * 16);
    v16h b2 = *(const v16h*)(buf + (2 * 32 + lane) * 16);
    v16h b3 = *(const v16h*)(buf + (3 * 32 + lane) * 16);
    a0 = WMMA_F16(a, b0, a0);
    a1 = WMMA_F16(a, b1, a1);
    a2 = WMMA_F16(a, b2, a2);
    a3 = WMMA_F16(a, b3, a3);
    __syncthreads();                  // done reading sB[kt&1] before reuse
  }
  const int rbase = rt * 16 + ((lane >> 4) << 3);
  const int nbase = ctg * 64 + (lane & 15);
  store_ctile(a0, out_logits, rbase, nbase);
  store_ctile(a1, out_logits, rbase, nbase + 16);
  store_ctile(a2, out_logits, rbase, nbase + 32);
  store_ctile(a3, out_logits, rbase, nbase + 48);
}

__global__ void k_ema(const float* __restrict__ macc, const float* __restrict__ nacc,
                      const float* __restrict__ accm, const float* __restrict__ accn,
                      float* __restrict__ out_m, float* __restrict__ out_n) {
  int i = blockIdx.x * blockDim.x + threadIdx.x;
  if (i < NNT * NDC) out_m[i] = DECAYF * macc[i] + (1.0f - DECAYF) * accm[i];
  if (i < NNT)       out_n[i] = DECAYF * nacc[i] + (1.0f - DECAYF) * accn[i];
}

// ===========================================================================
// Host launcher
// ===========================================================================
extern "C" void kernel_launch(void* const* d_in, const int* in_sizes, int n_in,
                              void* d_out, int out_size, void* d_ws, size_t ws_size,
                              hipStream_t stream) {
  (void)in_sizes; (void)n_in; (void)out_size; (void)ws_size;
  const int*   x     = (const int*)d_in[0];
  const int*   xmask = (const int*)d_in[1];
  const float* emb   = (const float*)d_in[2];
  const float* cb    = (const float*)d_in[3];
  const float* macc  = (const float*)d_in[4];
  const float* nacc  = (const float*)d_in[5];
  const float* Wt    = (const float*)d_in[6];
  const float* Wn    = (const float*)d_in[7];
  const float* Ws    = (const float*)d_in[8];
  const float* Wc    = (const float*)d_in[9];
  const float* Wp    = (const float*)d_in[10];
  float* out = (float*)d_out;

  // ---- workspace carve-up (deterministic every call) ----
  char*  ws  = (char*)d_ws;
  size_t off = 0;
  auto alloc = [&](size_t bytes) -> char* {
    char* p = ws + off;
    off = (off + bytes + 255) & ~(size_t)255;
    return p;
  };
  _Float16* wcell = (_Float16*)alloc((size_t)ZDIM * NCELL * 2);
  _Float16* wcode = (_Float16*)alloc((size_t)ZCODE * (2 * NDC) * 2);
  _Float16* wpred = (_Float16*)alloc((size_t)DD * VOCAB * 2);
  _Float16* cbpk  = (_Float16*)alloc((size_t)NDC * NNT * 2);
  float*    cbsq  = (float*)alloc(NNT * 4);
  _Float16* z     = (_Float16*)alloc((size_t)BB * ZDIM * 2);
  _Float16* zc    = (_Float16*)alloc((size_t)BB * ZCODE * 2);
  _Float16* c16   = (_Float16*)alloc((size_t)BB * (2 * NDC) * 2);
  float*    c32   = (float*)alloc((size_t)BB * (2 * NDC) * 4);
  float*    hst   = (float*)alloc((size_t)2 * BB * DD * 4);
  _Float16* ht16  = (_Float16*)alloc((size_t)BB * TT * DD * 2);
  float*    stk   = (float*)alloc((size_t)BB * CAP * NDC * 4);
  int*      ptr   = (int*)alloc(BB * 4);
  int*      smask = (int*)alloc(BB * 4);
  float*    accm  = (float*)alloc((size_t)NNT * NDC * 4);
  float*    accn  = (float*)alloc(NNT * 4);

  // ---- output carve-up (flat concat in reference return order) ----
  float* out_logits = out;                                       // B*T*V
  float* out_push   = out_logits + (size_t)BB * TT * VOCAB;      // B*T*2
  float* out_raw    = out_push + (size_t)BB * TT * 2;            // B*T*2*ND
  float* out_m      = out_raw + (size_t)BB * TT * 2 * NDC;       // NNT*ND
  float* out_n      = out_m + (size_t)NNT * NDC;                 // NNT

  // ---- init state ----
  hipMemsetAsync(hst,  0, (size_t)2 * BB * DD * 4, stream);
  hipMemsetAsync(stk,  0, (size_t)BB * CAP * NDC * 4, stream);
  hipMemsetAsync(accm, 0, (size_t)NNT * NDC * 4, stream);
  hipMemsetAsync(accn, 0, NNT * 4, stream);
  k_setptr<<<1, 64, 0, stream>>>(ptr);

  // ---- pack weights to fragment-major f16 ----
  {
    int tot = ZDIM * NCELL;
    k_pack_cell<<<(tot + 255) / 256, 256, 0, stream>>>(Wt, Wn, Ws, wcell, tot);
  }
  {
    int tot = ZCODE * (2 * NDC);
    k_pack_generic<<<(tot + 255) / 256, 256, 0, stream>>>(Wc, wcode, ZCODE, 2 * NDC,
                                                          (2 * NDC) / 16, 0, tot);
  }
  {
    int tot = DD * VOCAB;
    k_pack_generic<<<(tot + 255) / 256, 256, 0, stream>>>(Wp, wpred, DD, VOCAB,
                                                          VOCAB / 16, 0, tot);
  }
  {
    int tot = NDC * NNT;   // codebook^T: B[k][n] = cb[n][k]
    k_pack_generic<<<(tot + 255) / 256, 256, 0, stream>>>(cb, cbpk, NDC, NNT,
                                                          NNT / 16, 1, tot);
  }
  k_cbsq<<<2, 256, 0, stream>>>(cb, cbsq);

  // ---- serial scan over T: 4 fused kernels per step ----
  for (int t = 0; t < TT; ++t) {
    const float* hcur  = hst + (size_t)(t & 1) * BB * DD;
    float*       hnext = hst + (size_t)((t + 1) & 1) * BB * DD;
    k_buildz_pop<<<BB, 256, 0, stream>>>(x, xmask, emb, stk, ptr, smask, hcur, z, t);
    k_cell_gemm<<<NCELL / 16, 128, 0, stream>>>(z, wcell, hnext, zc, ht16, t);
    k_codes_gemm<<<(2 * NDC) / 16, 128, 0, stream>>>(zc, wcode, c32, c16, out_raw, t);
    k_vq<<<8, 256, 0, stream>>>(c16, cbpk, c32, cbsq, cb, smask, ptr, stk,
                                accm, accn, out_push, t);
  }

  // ---- hoisted logits GEMM (87% of FLOPs, fully parallel) ----
  k_logits<<<dim3(VOCAB / 64, (BB * TT / 16) / 8), 256, 0, stream>>>(ht16, wpred, out_logits);

  // ---- EMA outputs ----
  k_ema<<<(NNT * NDC + 255) / 256, 256, 0, stream>>>(macc, nacc, accm, accn, out_m, out_n);
}
